// HybridDenoisingRouter_33947421508032
// MI455X (gfx1250) — matrix operand order repaired and checked by reading, not compiled, not run on hardware
//
#include <hip/hip_runtime.h>
#include <hip/hip_bf16.h>

// ===========================================================================
// HybridDenoisingRouter for MI455X (gfx1250, wave32, WMMA + TDM + async LDS).
//
// 3x3 convs + 2x2 transposed convs are implicit GEMM on
// v_wmma_f32_16x16x32_f16 (f16 in, f32 accumulate).
//  - wave tile: 16(pix) x 32(ch)  -> 2 accumulators, 4 WMMA per 64-wide K-step
//  - A (im2col) tile staged by VALU with an incremental coordinate walk
//    (one divide per K-step), 16B vector LDS stores
//  - B (weights) tile (32 x 64 f16, N-major, zero-padded):
//      * conv kernel:   TENSOR_LOAD_TO_LDS (TDM, TENSORcnt) with LDS pad
//                       feature matching the 144B row pitch
//      * convT2 kernel: GLOBAL_LOAD_ASYNC_TO_LDS_B128 (ASYNCcnt)
//  - fragments read back as ds_load_b128 (pitch 72 f16 keeps 16B alignment)
// Activations held in f16 between layers (~300MB total traffic ~= 13us at
// 23.3 TB/s, same order as ~70 GFLOP of math at WMMA rate -> balanced).
// ===========================================================================

typedef __attribute__((ext_vector_type(16))) _Float16 v16h;
typedef __attribute__((ext_vector_type(8)))  _Float16 v8h;
typedef __attribute__((ext_vector_type(8)))  float    v8f;
typedef __attribute__((ext_vector_type(4)))  unsigned v4u;
typedef __attribute__((ext_vector_type(8)))  int      v8i;
typedef __attribute__((ext_vector_type(4)))  int      v4i;

union AF { v16h v; v8h q[2]; _Float16 h[16]; };
union CF { v8f  v; float    f[8];  };
union H8 { v8h  v; _Float16 h[8];  };

__device__ __forceinline__ float gelu_f(float x) {
  return 0.5f * x * (1.0f + erff(x * 0.70710678118654752f));
}

// ---------------------------------------------------------------------------
// weight repack: -> N-major f16, zero-padded to Npad x Kpad.
// n_major_src=1: src is w[n*K + k]   (conv OIHW flat, N=Cout, K=Cin*9)
// n_major_src=0: src is w[k*N + n]   (convT2 (in,out,2,2) flat, K=Cin, N=Cout*4)
// ---------------------------------------------------------------------------
__global__ void k_pack_w(const float* __restrict__ w, _Float16* __restrict__ out,
                         int N, int K, int Npad, int Kpad, int n_major_src) {
  int i = blockIdx.x * 256 + threadIdx.x;
  if (i >= Npad * Kpad) return;
  int n = i / Kpad, k = i - n * Kpad;
  float v = 0.0f;
  if (n < N && k < K) v = n_major_src ? w[n * K + k] : w[k * N + n];
  out[i] = (_Float16)v;
}

__global__ void k_cvt_f32_to_f16(const float* __restrict__ in,
                                 _Float16* __restrict__ out, int n) {
  int i = blockIdx.x * 256 + threadIdx.x;
  if (i < n) out[i] = (_Float16)in[i];
}

// ---------------------------------------------------------------------------
// TDM B-tile stage: one TENSOR_LOAD_TO_LDS moves 32 rows x 128B (N-major,
// row stride Kpad*2 bytes) into LDS with a 16B pad after every 128B row
// (pad_interval code 4 -> 128B, pad_amount code 3 -> 16B) => 144B LDS pitch.
// D# layout per CDNA5 ISA.
// 6-arg builtin form (clang-23 / therock-10.0 headers).
// ---------------------------------------------------------------------------
__device__ __forceinline__ void stage_B_tdm(const _Float16* wNK, int Kpad,
                                            int nbase, int kc,
                                            _Float16 (*Bs)[72]) {
  unsigned long long ga = (unsigned long long)(uintptr_t)wNK +
                          (unsigned long long)(((long)nbase * Kpad + kc) * 2);
  unsigned lds = (unsigned)(uintptr_t)&Bs[0][0];

  v4u g0;
  g0[0] = 1u;                                   // count=1 (valid user D#)
  g0[1] = lds;                                  // lds_addr (bytes)
  g0[2] = (unsigned)ga;                         // global_addr[31:0]
  g0[3] = (unsigned)(ga >> 32) | (2u << 30);    // global_addr[56:32] | type=2

  const unsigned td0   = 1u << 20;              // tensor_dim0 (large: no clip)
  const unsigned td1   = 1u << 20;              // tensor_dim1
  const unsigned tile0 = 16;                    // 128B row / 8B units
  const unsigned tile1 = 32;                    // 32 rows
  const unsigned str0  = (unsigned)(Kpad / 4);  // row stride in 8B units

  v8i g1;
  g1[0] = (int)((3u << 16) |                    // data_size = 3 (8B)
                (1u << 20) |                    // pad_enable
                (4u << 22) |                    // pad_interval: 2^4*8B = 128B
                (3u << 25));                    // pad_amount: 4 DWORDs = 16B
  g1[1] = (int)((td0 & 0xFFFFu) << 16);                  // tensor_dim0[15:0]
  g1[2] = (int)((td0 >> 16) | ((td1 & 0xFFFFu) << 16));  // td0[31:16]|td1[15:0]
  g1[3] = (int)((td1 >> 16) | (tile0 << 16));            // td1[31:16]|tile_dim0
  g1[4] = (int)tile1;                                    // tile_dim1, tile_dim2=0
  g1[5] = (int)str0;                                     // tensor_dim0_stride lo
  g1[6] = 0;                                             // stride hi | d1_stride
  g1[7] = 0;
  v4i z4 = {0, 0, 0, 0};
  v8i z8 = {0, 0, 0, 0, 0, 0, 0, 0};
  __builtin_amdgcn_tensor_load_to_lds(g0, g1, z4, z4, z8, 0);
  __builtin_amdgcn_s_wait_tensorcnt(0);
}

// ---------------------------------------------------------------------------
// async B-tile stage (convT2): 32 rows x 128B into LDS via per-lane 16B
// GLOBAL_LOAD_ASYNC_TO_LDS_B128, waves 0-1, tracked on ASYNCcnt.
// ---------------------------------------------------------------------------
__device__ __forceinline__ void stage_B_async(unsigned long long wbase,
                                              int Kpad, int nbase, int kc,
                                              _Float16 (*Bs)[72],
                                              int lane, int wid) {
  #pragma unroll
  for (int i = 0; i < 4; ++i) {
    int flat = wid * 2048 + i * 512 + lane * 16;  // byte index into 4KB tile
    int row  = flat >> 7;                         // 128B per N-row
    int col  = flat & 127;
    unsigned lds  = (unsigned)(uintptr_t)&Bs[row][0] + (unsigned)col;
    unsigned voff = (unsigned)(((nbase + row) * Kpad + kc) * 2 + col);
    asm volatile("global_load_async_to_lds_b128 %0, %1, %2"
                 :: "v"(lds), "v"(voff), "s"(wbase) : "memory");
  }
  asm volatile("s_wait_asynccnt 0x0" ::: "memory");
}

// ---------------------------------------------------------------------------
// 3x3 conv (pad=1, stride 1|2) implicit GEMM.  Dual-input channel split
// (ci < Cin1 -> in1 else in2) implements skip-concats without copies.
// Block = 128 thr = 4 waves; wave tile 16(pix) x 32(ch); K-step 64.
// ---------------------------------------------------------------------------
__global__ __launch_bounds__(128)
void k_conv3x3_wmma(const _Float16* __restrict__ in1,
                    const _Float16* __restrict__ in2,
                    int Cin1, int Cin, int Hin, int Win, int stride,
                    const _Float16* __restrict__ wNK, int Kpad,
                    const float* __restrict__ bias,
                    int Cout, _Float16* __restrict__ out)
{
  const int Hout = Hin / stride, Wout = Win / stride;
  const int HWo  = Hout * Wout;
  const int HWi  = Hin * Win;
  const int K    = Cin * 9;
  const int Cin2 = Cin - Cin1;

  __shared__ _Float16 As[64][72];   // 64 pix x 64 K  (144B rows, 16B aligned)
  __shared__ _Float16 Bs[32][72];   // 32 ch  x 64 K  (N-major)

  const int tid  = threadIdx.x;
  const int lane = tid & 31, wid = tid >> 5, half = lane >> 4;
  const int pbase = blockIdx.x * 64, cbase = blockIdx.y * 32;

  // staging coords: one A row + 32-wide K chunk per thread; decode pixel once.
  const int srow = tid >> 1;
  const int skb  = (tid & 1) * 32;
  const int spm  = pbase + srow;
  const int sb   = spm / HWo;
  const int srem = spm - sb * HWo;
  const int soh  = srem / Wout, sow = srem - soh * Wout;
  const int ih0  = soh * stride - 1, iw0 = sow * stride - 1;

  CF acc0, acc1;
  #pragma unroll
  for (int i = 0; i < 8; ++i) { acc0.f[i] = 0.0f; acc1.f[i] = 0.0f; }

  for (int kc = 0; kc < K; kc += 64) {
    // ---- A stage: incremental im2col walk (one divide per K-step) ----
    {
      int k  = kc + skb;
      int ci = k / 9;
      int r  = k - ci * 9;
      int kh = r / 3;
      int kw = r - kh * 3;
      int ih = ih0 + kh, iw = iw0 + kw;
      int rowoff = ih * Win;
      const _Float16* cur =
          (ci < Cin1) ? in1 + ((long)sb * Cin1 + ci) * HWi
                      : in2 + ((long)sb * Cin2 + (ci - Cin1)) * HWi;
      #pragma unroll
      for (int g2 = 0; g2 < 4; ++g2) {
        H8 pk;
        #pragma unroll
        for (int e = 0; e < 8; ++e) {
          bool ok = (k < K) & ((unsigned)ih < (unsigned)Hin) &
                    ((unsigned)iw < (unsigned)Win);
          pk.h[e] = ok ? cur[rowoff + iw] : (_Float16)0.0f;
          // advance (kw,kh,ci) with carried wraps -> selects after unroll
          ++k; ++kw; ++iw;
          if (kw == 3) {
            kw = 0; iw = iw0;
            ++kh; ++ih; rowoff += Win;
            if (kh == 3) {
              kh = 0; ih = ih0; rowoff = ih0 * Win;
              ++ci;
              cur = (ci == Cin1) ? in2 + (long)sb * Cin2 * HWi : cur + HWi;
            }
          }
        }
        *(v8h*)&As[srow][skb + g2 * 8] = pk.v;
      }
    }
    // ---- B stage: one TDM descriptor moves the whole 32x64 tile ----
    if (wid == 0) stage_B_tdm(wNK, Kpad, cbase, kc, Bs);
    __syncthreads();

    // ---- 4 WMMA per K-step (2 K sub-tiles x 2 N tiles) ----
    const int mrow = (wid << 4) + (lane & 15);
    const int ncol = lane & 15;
    #pragma unroll
    for (int s = 0; s < 2; ++s) {
      AF a, b0, b1;
      a.q[0]  = *(const v8h*)&As[mrow][s * 32 + half * 16];
      a.q[1]  = *(const v8h*)&As[mrow][s * 32 + half * 16 + 8];
      b0.q[0] = *(const v8h*)&Bs[ncol][s * 32 + half * 16];
      b0.q[1] = *(const v8h*)&Bs[ncol][s * 32 + half * 16 + 8];
      b1.q[0] = *(const v8h*)&Bs[16 + ncol][s * 32 + half * 16];
      b1.q[1] = *(const v8h*)&Bs[16 + ncol][s * 32 + half * 16 + 8];
      acc0.v = __builtin_amdgcn_wmma_f32_16x16x32_f16(
          false, a.v, false, b0.v, (short)0, acc0.v, false, false);
      acc1.v = __builtin_amdgcn_wmma_f32_16x16x32_f16(
          false, a.v, false, b1.v, (short)0, acc1.v, false, false);
    }
    __syncthreads();
  }

  // ---- epilogue: D layout (VGPR r -> M = r + 8*half, N = lane&15) ----
  const int ncol = lane & 15;
  #pragma unroll
  for (int j = 0; j < 2; ++j) {
    int ch = cbase + j * 16 + ncol;
    if (ch < Cout) {
      const float bv = bias[ch];
      const CF* ac = j ? &acc1 : &acc0;
      #pragma unroll
      for (int r = 0; r < 8; ++r) {
        int pm  = pbase + (wid << 4) + half * 8 + r;
        int b   = pm / HWo;
        int rem = pm - b * HWo;
        int oh  = rem / Wout, ow = rem - oh * Wout;
        out[(((long)b * Cout + ch) * Hout + oh) * (long)Wout + ow] =
            (_Float16)(ac->f[r] + bv);
      }
    }
  }
}

// ---------------------------------------------------------------------------
// ConvTranspose2d k=2,s=2 as a plain GEMM: M = input pixels, K = Cin,
// N = Cout*4 (co,i,j); 2x2 scatter epilogue.  Wave tile 16 x 32.
// ---------------------------------------------------------------------------
__global__ __launch_bounds__(128)
void k_convt2_wmma(const _Float16* __restrict__ x,
                   const _Float16* __restrict__ wNK,
                   const float* __restrict__ bias,
                   _Float16* __restrict__ out,
                   int Cin, int Hh, int Ww, int Cout)
{
  const int HWi  = Hh * Ww;
  const int Hout = Hh * 2, Wout = Ww * 2;

  __shared__ _Float16 As[64][72];
  __shared__ _Float16 Bs[32][72];

  const int tid  = threadIdx.x;
  const int lane = tid & 31, wid = tid >> 5, half = lane >> 4;
  const int pbase = blockIdx.x * 64, nbase = blockIdx.y * 32;

  const int srow = tid >> 1;
  const int skb  = (tid & 1) * 32;
  const int spm  = pbase + srow;
  const int sb   = spm / HWi;
  const int spix = spm - sb * HWi;

  CF acc0, acc1;
  #pragma unroll
  for (int i = 0; i < 8; ++i) { acc0.f[i] = 0.0f; acc1.f[i] = 0.0f; }

  const unsigned long long wbase = (unsigned long long)(uintptr_t)wNK;

  for (int kc = 0; kc < Cin; kc += 64) {
    {
      int k = kc + skb;
      const _Float16* cur = x + ((long)sb * Cin + k) * HWi + spix;
      #pragma unroll
      for (int g2 = 0; g2 < 4; ++g2) {
        H8 pk;
        #pragma unroll
        for (int e = 0; e < 8; ++e) {
          pk.h[e] = (k < Cin) ? *cur : (_Float16)0.0f;
          ++k; cur += HWi;
        }
        *(v8h*)&As[srow][skb + g2 * 8] = pk.v;
      }
    }
    if (wid < 2) stage_B_async(wbase, Cin, nbase, kc, Bs, lane, wid);
    __syncthreads();

    const int mrow = (wid << 4) + (lane & 15);
    const int ncol = lane & 15;
    #pragma unroll
    for (int s = 0; s < 2; ++s) {
      AF a, b0, b1;
      a.q[0]  = *(const v8h*)&As[mrow][s * 32 + half * 16];
      a.q[1]  = *(const v8h*)&As[mrow][s * 32 + half * 16 + 8];
      b0.q[0] = *(const v8h*)&Bs[ncol][s * 32 + half * 16];
      b0.q[1] = *(const v8h*)&Bs[ncol][s * 32 + half * 16 + 8];
      b1.q[0] = *(const v8h*)&Bs[16 + ncol][s * 32 + half * 16];
      b1.q[1] = *(const v8h*)&Bs[16 + ncol][s * 32 + half * 16 + 8];
      acc0.v = __builtin_amdgcn_wmma_f32_16x16x32_f16(
          false, a.v, false, b0.v, (short)0, acc0.v, false, false);
      acc1.v = __builtin_amdgcn_wmma_f32_16x16x32_f16(
          false, a.v, false, b1.v, (short)0, acc1.v, false, false);
    }
    __syncthreads();
  }

  const int ncol = lane & 15;
  #pragma unroll
  for (int j = 0; j < 2; ++j) {
    const int n  = nbase + j * 16 + ncol;
    const int co = n >> 2, oi = (n >> 1) & 1, oj = n & 1;
    const float bv = bias[co];
    const CF* ac = j ? &acc1 : &acc0;
    #pragma unroll
    for (int r = 0; r < 8; ++r) {
      int pm  = pbase + (wid << 4) + half * 8 + r;
      int b   = pm / HWi;
      int pix = pm - b * HWi;
      int h   = pix / Ww, w = pix - h * Ww;
      out[(((long)b * Cout + co) * Hout + (2 * h + oi)) * (long)Wout + (2 * w + oj)] =
          (_Float16)(ac->f[r] + bv);
    }
  }
}

// ---------------------------------------------------------------------------
// GroupNorm + GELU fused; one block per (batch, group); f32 stats; v8h I/O.
// ---------------------------------------------------------------------------
__global__ __launch_bounds__(256)
void k_gn_gelu(const _Float16* __restrict__ in, _Float16* __restrict__ out,
               const float* __restrict__ gamma, const float* __restrict__ beta,
               int C, int groups, int HW)
{
  __shared__ float rs[256], rss[256];
  __shared__ float s_mean, s_inv;

  const int b   = blockIdx.x / groups;
  const int g   = blockIdx.x - b * groups;
  const int cpg = C / groups;
  const long Ng = (long)cpg * HW;
  const long base = ((long)b * C + (long)g * cpg) * HW;
  const int tid = threadIdx.x;

  float s = 0.0f, ss = 0.0f;
  for (long i = (long)tid * 8; i < Ng; i += 2048) {
    H8 xv; xv.v = *(const v8h*)&in[base + i];
    #pragma unroll
    for (int e = 0; e < 8; ++e) {
      float x = (float)xv.h[e];
      s += x; ss += x * x;
    }
  }
  rs[tid] = s; rss[tid] = ss;
  __syncthreads();
  for (int st = 128; st > 0; st >>= 1) {
    if (tid < st) { rs[tid] += rs[tid + st]; rss[tid] += rss[tid + st]; }
    __syncthreads();
  }
  if (tid == 0) {
    float mean = rs[0] / (float)Ng;
    float var  = rss[0] / (float)Ng - mean * mean;
    s_mean = mean;
    s_inv  = rsqrtf(var + 1e-5f);
  }
  __syncthreads();
  const float mean = s_mean, inv = s_inv;

  for (long i = (long)tid * 8; i < Ng; i += 2048) {
    int c = g * cpg + (int)(i / HW);           // HW multiple of 8 -> uniform in chunk
    float ga = gamma[c], be = beta[c];
    H8 xv, yv;
    xv.v = *(const v8h*)&in[base + i];
    #pragma unroll
    for (int e = 0; e < 8; ++e) {
      float x = (float)xv.h[e];
      yv.h[e] = (_Float16)gelu_f((x - mean) * inv * ga + be);
    }
    *(v8h*)&out[base + i] = yv.v;
  }
}

// ---------------------------------------------------------------------------
// na_out 1x1 (32->1) + sigmoid + expert gating + fusion-input build.
// ---------------------------------------------------------------------------
__global__ void k_na_gate(const _Float16* __restrict__ d2,
                          const float* __restrict__ w, const float* __restrict__ b,
                          const float* __restrict__ naf, const float* __restrict__ diff,
                          float* __restrict__ gated, _Float16* __restrict__ fus_in,
                          int n, int HW)
{
  int i = blockIdx.x * 256 + threadIdx.x;
  if (i >= n) return;
  int bi  = i / HW;
  int pix = i - bi * HW;
  float dot = b[0];
  #pragma unroll
  for (int c = 0; c < 32; ++c)
    dot += w[c] * (float)d2[((long)bi * 32 + c) * (long)HW + pix];
  float mask = 1.0f / (1.0f + expf(-dot));
  float nv = naf[i], dv = diff[i];
  gated[i] = mask * dv + (1.0f - mask) * nv;
  long fb = (long)bi * 3 * HW + pix;
  fus_in[fb]          = (_Float16)nv;
  fus_in[fb + HW]     = (_Float16)dv;
  fus_in[fb + 2 * HW] = (_Float16)mask;
}

// ---------------------------------------------------------------------------
// fus_out 1x1 (24->1) + residual add with gated expert sum -> d_out (f32).
// ---------------------------------------------------------------------------
__global__ void k_fus_out(const _Float16* __restrict__ f2,
                          const float* __restrict__ w, const float* __restrict__ b,
                          const float* __restrict__ gated,
                          float* __restrict__ out, int n, int HW)
{
  int i = blockIdx.x * 256 + threadIdx.x;
  if (i >= n) return;
  int bi  = i / HW;
  int pix = i - bi * HW;
  float dot = b[0];
  #pragma unroll
  for (int c = 0; c < 24; ++c)
    dot += w[c] * (float)f2[((long)bi * 24 + c) * (long)HW + pix];
  out[i] = gated[i] + dot;
}

// ===========================================================================
// Orchestration
// ===========================================================================
extern "C" void kernel_launch(void* const* d_in, const int* in_sizes, int n_in,
                              void* d_out, int out_size, void* d_ws, size_t ws_size,
                              hipStream_t stream)
{
  // positional inputs, then params pytree leaves in sorted-key order
  // (jax tree_leaves convention): per-cblk keys sort as b, be, g, w.
  const float* noisy = (const float*)d_in[0];
  const float* naf   = (const float*)d_in[1];
  const float* diff  = (const float*)d_in[2];
  enum {
    DEC2_B = 3,  DEC2_BE, DEC2_G, DEC2_W,
    DEC3_B = 7,  DEC3_BE, DEC3_G, DEC3_W,
    ENC1_B = 11, ENC1_BE, ENC1_G, ENC1_W,
    ENC2_B = 15, ENC2_BE, ENC2_G, ENC2_W,
    ENC3_B = 19, ENC3_BE, ENC3_G, ENC3_W,
    FUS1_B = 23, FUS1_BE, FUS1_G, FUS1_W,
    FUS2_B = 27, FUS2_BE, FUS2_G, FUS2_W,
    FUSO_B = 31, FUSO_W = 32,
    MID_B  = 33, MID_BE, MID_G, MID_W,
    NAO_B  = 37, NAO_W = 38,
    UP2_B  = 39, UP2_W = 40,
    UP3_B  = 41, UP3_W = 42,
  };
  auto F = [&](int i) { return (const float*)d_in[i]; };

  const int B = 8, HW = 65536;

  size_t off = 0;
  auto alloc = [&](size_t elems, size_t esz) -> void* {
    void* p = (char*)d_ws + off;
    off += ((elems * esz + 255) / 256) * 256;
    return p;
  };
  _Float16* x16    = (_Float16*)alloc(524288, 2);
  // padded N-major weights (Npad x Kpad); Npad mult of 32, Kpad mult of 64
  _Float16* w_enc1 = (_Float16*)alloc(32 * 64, 2);
  _Float16* w_enc2 = (_Float16*)alloc(64 * 320, 2);
  _Float16* w_enc3 = (_Float16*)alloc(128 * 576, 2);
  _Float16* w_mid  = (_Float16*)alloc(128 * 1152, 2);
  _Float16* w_dec3 = (_Float16*)alloc(64 * 1152, 2);
  _Float16* w_dec2 = (_Float16*)alloc(32 * 576, 2);
  _Float16* w_fus1 = (_Float16*)alloc(64 * 64, 2);
  _Float16* w_fus2 = (_Float16*)alloc(32 * 448, 2);
  _Float16* w_up3  = (_Float16*)alloc(256 * 128, 2);
  _Float16* w_up2  = (_Float16*)alloc(128 * 64, 2);
  _Float16* raw    = (_Float16*)alloc(25165824, 2);  // largest pre-GN tensor (f1)
  _Float16* e1     = (_Float16*)alloc(16777216, 2);
  _Float16* e2     = (_Float16*)alloc(8388608, 2);
  _Float16* e3     = (_Float16*)alloc(4194304, 2);
  _Float16* mbuf   = (_Float16*)alloc(4194304, 2);
  _Float16* d3up   = (_Float16*)alloc(8388608, 2);
  _Float16* d3     = (_Float16*)alloc(8388608, 2);
  _Float16* d2up   = (_Float16*)alloc(16777216, 2);
  _Float16* d2     = (_Float16*)alloc(16777216, 2);
  _Float16* fusin  = (_Float16*)alloc(1572864, 2);
  _Float16* f1     = (_Float16*)alloc(25165824, 2);
  _Float16* f2b    = (_Float16*)alloc(12582912, 2);
  float*    gated  = (float*)alloc(524288, 4);

  auto g1 = [](int n) { return dim3((n + 255) / 256); };
  auto pack = [&](const float* w, _Float16* o, int N, int K, int Npad, int Kpad,
                  int nmaj) {
    k_pack_w<<<g1(Npad * Kpad), 256, 0, stream>>>(w, o, N, K, Npad, Kpad, nmaj);
  };

  k_cvt_f32_to_f16<<<g1(524288), 256, 0, stream>>>(noisy, x16, 524288);
  pack(F(ENC1_W), w_enc1, 32, 9,     32, 64,   1);
  pack(F(ENC2_W), w_enc2, 64, 288,   64, 320,  1);
  pack(F(ENC3_W), w_enc3, 128, 576,  128, 576, 1);
  pack(F(MID_W),  w_mid,  128, 1152, 128, 1152,1);
  pack(F(DEC3_W), w_dec3, 64, 1152,  64, 1152, 1);
  pack(F(DEC2_W), w_dec2, 32, 576,   32, 576,  1);
  pack(F(FUS1_W), w_fus1, 48, 27,    64, 64,   1);
  pack(F(FUS2_W), w_fus2, 24, 432,   32, 448,  1);
  pack(F(UP3_W),  w_up3,  256, 128,  256, 128, 0);
  pack(F(UP2_W),  w_up2,  128, 64,   128, 64,  0);

  auto conv = [&](const _Float16* in1, const _Float16* in2, int Cin1, int Cin,
                  int Hin, int stride, const _Float16* wkn, int Kpad,
                  const float* bias, int Cout, _Float16* o) {
    int Mtot = B * (Hin / stride) * (Hin / stride);
    dim3 grid(Mtot / 64, (Cout + 31) / 32);
    k_conv3x3_wmma<<<grid, 128, 0, stream>>>(in1, in2, Cin1, Cin, Hin, Hin,
                                             stride, wkn, Kpad, bias, Cout, o);
  };
  auto gng = [&](const _Float16* in, _Float16* o, const float* g_, const float* be,
                 int C, int groups, int hw) {
    k_gn_gelu<<<dim3(B * groups), 256, 0, stream>>>(in, o, g_, be, C, groups, hw);
  };

  // ---- NoiseAnalyzer encoder ----
  conv(x16, x16, 1, 1, 256, 1, w_enc1, 64, F(ENC1_B), 32, raw);
  gng(raw, e1, F(ENC1_G), F(ENC1_BE), 32, 8, 65536);          // e1 [8,32,256,256]
  conv(e1, e1, 32, 32, 256, 2, w_enc2, 320, F(ENC2_B), 64, raw);
  gng(raw, e2, F(ENC2_G), F(ENC2_BE), 64, 8, 16384);          // e2 [8,64,128,128]
  conv(e2, e2, 64, 64, 128, 2, w_enc3, 576, F(ENC3_B), 128, raw);
  gng(raw, e3, F(ENC3_G), F(ENC3_BE), 128, 8, 4096);          // e3 [8,128,64,64]
  conv(e3, e3, 128, 128, 64, 1, w_mid, 1152, F(MID_B), 128, raw);
  gng(raw, mbuf, F(MID_G), F(MID_BE), 128, 8, 4096);          // mid

  // ---- decoder (skip-concat via dual-input conv) ----
  k_convt2_wmma<<<dim3((B * 64 * 64) / 64, 8), 128, 0, stream>>>(
      mbuf, w_up3, F(UP3_B), d3up, 128, 64, 64, 64);          // d3up [8,64,128,128]
  conv(d3up, e2, 64, 128, 128, 1, w_dec3, 1152, F(DEC3_B), 64, raw);
  gng(raw, d3, F(DEC3_G), F(DEC3_BE), 64, 8, 16384);          // d3
  k_convt2_wmma<<<dim3((B * 128 * 128) / 64, 4), 128, 0, stream>>>(
      d3, w_up2, F(UP2_B), d2up, 64, 128, 128, 32);           // d2up [8,32,256,256]
  conv(d2up, e1, 32, 64, 256, 1, w_dec2, 576, F(DEC2_B), 32, raw);
  gng(raw, d2, F(DEC2_G), F(DEC2_BE), 32, 8, 65536);          // d2

  // ---- mask + gating + fusion input ----
  k_na_gate<<<g1(B * HW), 256, 0, stream>>>(d2, F(NAO_W), F(NAO_B), naf, diff,
                                            gated, fusin, B * HW, HW);

  // ---- fusion head ----
  conv(fusin, fusin, 3, 3, 256, 1, w_fus1, 64, F(FUS1_B), 48, raw);
  gng(raw, f1, F(FUS1_G), F(FUS1_BE), 48, 8, 65536);
  conv(f1, f1, 48, 48, 256, 1, w_fus2, 448, F(FUS2_B), 24, raw);
  gng(raw, f2b, F(FUS2_G), F(FUS2_BE), 24, 4, 65536);

  // ---- final 1x1 + residual ----
  k_fus_out<<<g1(B * HW), 256, 0, stream>>>(f2b, F(FUSO_W), F(FUSO_B), gated,
                                            (float*)d_out, B * HW, HW);
}
